// TemporalSelfAttention_30829275251288
// MI455X (gfx1250) — matrix operand order repaired and checked by reading, not compiled, not run on hardware
//
#include <hip/hip_runtime.h>
#include <hip/hip_bf16.h>
#include <math.h>

// ---------------------------------------------------------------------------
// TemporalSelfAttention on MI455X (gfx1250, wave32, WMMA)
// All GEMM-like stages use v_wmma_f32_16x16x32_bf16.
// Weights and V are pre-transposed so every WMMA operand loads as contiguous
// 16B vectors (global_load_b128) instead of strided u16 loads.
// ---------------------------------------------------------------------------

typedef __bf16 bf16_t;
typedef __attribute__((ext_vector_type(16))) __bf16 v16bf;
typedef __attribute__((ext_vector_type(8)))  __bf16 v8bf;
typedef __attribute__((ext_vector_type(8)))  float  v8f;

#define T_SEQ   1024
#define E_DIM   256
#define HEADS   8
#define HD      32          // head dim == WMMA K for bf16
#define BN      8           // B * N
#define QKV_N   768

// ---- WMMA operand loaders (contiguous k = 16g+i, same order in A and B) -------

// A-matrix 16x32 bf16: lane holds row (lane%16); halves i = K 16g+i
// -> two contiguous 16B loads.
__device__ __forceinline__ v16bf load_a_contig(const bf16_t* row, int g) {
    v8bf lo = *(const v8bf*)(row + 16 * g);
    v8bf hi = *(const v8bf*)(row + 16 * g + 8);
    v16bf a;
#pragma unroll
    for (int i = 0; i < 8; ++i) { a[i] = lo[i]; a[i + 8] = hi[i]; }
    return a;
}

// B-matrix 32x16 bf16: lane holds column (lane%16); halves i = K 16g+i.
// Caller passes a pointer where those 16 K-values are contiguous.
__device__ __forceinline__ v16bf load_b_contig16(const bf16_t* p) {
    v8bf lo = *(const v8bf*)(p);
    v8bf hi = *(const v8bf*)(p + 8);
    v16bf b;
#pragma unroll
    for (int i = 0; i < 8; ++i) { b[i] = lo[i]; b[i + 8] = hi[i]; }
    return b;
}

__device__ __forceinline__ float rowmax16(float v) {
#pragma unroll
    for (int m = 1; m < 16; m <<= 1) v = fmaxf(v, __shfl_xor(v, m, 32));
    return v;
}
__device__ __forceinline__ float rowsum16(float v) {
#pragma unroll
    for (int m = 1; m < 16; m <<= 1) v += __shfl_xor(v, m, 32);
    return v;
}

// ---- Stage 0: permute (B,T,N,E) -> (BN,T,E) and convert to bf16 ------------
__global__ void permute_input_kernel(const float* __restrict__ in,
                                     bf16_t* __restrict__ X) {
    int idx = blockIdx.x * blockDim.x + threadIdx.x;   // BN*T*E = 2^21 exact
    int e  = idx & 255;
    int t  = (idx >> 8) & 1023;
    int bn = idx >> 18;
    int b = bn >> 2, n = bn & 3;
    X[idx] = (bf16_t)in[((b * T_SEQ + t) * 4 + n) * E_DIM + e];
}

// Transpose-convert: out (cols x rows) row-major = in (rows x cols) transposed.
// out[n*rows + k] = in[k*cols + n]; writes contiguous.
__global__ void transpose_to_bf16_kernel(const float* __restrict__ in,
                                         bf16_t* __restrict__ out,
                                         int rows, int cols) {
    int idx = blockIdx.x * blockDim.x + threadIdx.x;
    if (idx >= rows * cols) return;
    int k = idx % rows;
    int n = idx / rows;
    out[idx] = (bf16_t)in[k * cols + n];
}

// ---- Stage 1: QKV GEMM, C(8192x768) = X(8192x256) @ W(256x768) + bias ------
// Wt is W transposed: (768 x 256) row-major -> B loads fully contiguous.
__global__ void qkv_gemm_kernel(const bf16_t* __restrict__ X,
                                const bf16_t* __restrict__ Wt,
                                const float*  __restrict__ bias,
                                bf16_t* __restrict__ out) {
    const int lane = threadIdx.x;
    const int l16  = lane & 15;
    const int g    = lane >> 4;
    const int tm   = blockIdx.x;     // 512
    const int tn   = blockIdx.y;     // 48
    const int col  = tn * 16 + l16;
    const bf16_t* arow = X + (tm * 16 + l16) * E_DIM;
    const bf16_t* brow = Wt + col * E_DIM;      // row of W^T = column of W
    v8f c = {};
#pragma unroll
    for (int kb = 0; kb < E_DIM; kb += 32) {
        v16bf a = load_a_contig(arow + kb, g);
        v16bf b = load_b_contig16(brow + kb + 16 * g);
        c = __builtin_amdgcn_wmma_f32_16x16x32_bf16(false, a, false, b,
                                                    (short)0, c, false, false);
    }
    const float bv = bias[col];
#pragma unroll
    for (int r = 0; r < 8; ++r) {
        int orow = tm * 16 + r + 8 * g;
        out[orow * QKV_N + col] = (bf16_t)(c[r] + bv);
    }
}

// ---- Stage 2: RoPE on q,k + split; V stored transposed per head ------------
// Q,K: (bnh, t, d) row-major.  V: (bnh, d, t) row-major (transposed).
__global__ void rope_split_kernel(const bf16_t* __restrict__ qkv,
                                  bf16_t* __restrict__ Qo,
                                  bf16_t* __restrict__ Ko,
                                  bf16_t* __restrict__ Vt) {
    int idx = blockIdx.x * blockDim.x + threadIdx.x;   // BN*T*H*16 = 2^20
    int j  = idx & 15;
    int h  = (idx >> 4) & 7;
    int t  = (idx >> 7) & 1023;
    int bn = idx >> 17;
    const bf16_t* row = qkv + (bn * T_SEQ + t) * QKV_N;
    float inv_freq = __powf(10000.0f, -(float)j * (1.0f / 16.0f));
    float ang = (float)t * inv_freq;
    float sn, cs;
    __sincosf(ang, &sn, &cs);
    const int cq   = h * HD;
    const int bnh  = bn * HEADS + h;
    const int base = (bnh * T_SEQ + t) * HD;
    float q1 = (float)row[cq + j],        q2 = (float)row[cq + 16 + j];
    Qo[base + j]      = (bf16_t)(q1 * cs - q2 * sn);
    Qo[base + 16 + j] = (bf16_t)(q2 * cs + q1 * sn);
    float k1 = (float)row[256 + cq + j],  k2 = (float)row[256 + cq + 16 + j];
    Ko[base + j]      = (bf16_t)(k1 * cs - k2 * sn);
    Ko[base + 16 + j] = (bf16_t)(k2 * cs + k1 * sn);
    // V transposed: Vt[bnh][d][t]
    bf16_t* vth = Vt + bnh * HD * T_SEQ;
    vth[j * T_SEQ + t]        = row[512 + cq + j];
    vth[(j + 16) * T_SEQ + t] = row[512 + cq + 16 + j];
}

// ---- Stage 3: causal flash attention, one wave per (bn,h,16-row q tile) ----
// scores: A = Q(16x32), B = K^T(32x16) -> one WMMA per 16-key tile.
// PV:     A = P(16x32 over 32 keys, staged via LDS), B = V(32x16) per d-half,
//         read from V^T so each lane's 16 keys are contiguous.
__global__ void attn_kernel(const bf16_t* __restrict__ Q,
                            const bf16_t* __restrict__ K,
                            const bf16_t* __restrict__ Vt,
                            bf16_t* __restrict__ AO) {
    const int lane = threadIdx.x;
    const int l16  = lane & 15;
    const int g    = lane >> 4;
    const int bnh  = blockIdx.x;       // bn*8 + h (64)
    const int q0   = blockIdx.y * 16;  // 64 tiles
    const bf16_t* Qh  = Q  + bnh * T_SEQ * HD;
    const bf16_t* Kh  = K  + bnh * T_SEQ * HD;
    const bf16_t* Vth = Vt + bnh * HD * T_SEQ;   // (d, t) layout

    __shared__ __align__(16) bf16_t Plds[16 * 32];

    const v16bf aq = load_a_contig(Qh + (q0 + l16) * HD, g);

    v8f O0 = {}, O1 = {};
    float mrow[8], lrow[8];
#pragma unroll
    for (int r = 0; r < 8; ++r) { mrow[r] = -INFINITY; lrow[r] = 0.0f; }

    const float scale  = 0.17677669529663687f;   // 1/sqrt(32)
    const int   nchunk = (q0 + 16 + 31) >> 5;    // 32 keys per chunk

    for (int ch = 0; ch < nchunk; ++ch) {
        const int s0 = ch * 32;
        v16bf bk0 = load_b_contig16(Kh + (s0 + l16) * HD + 16 * g);
        v16bf bk1 = load_b_contig16(Kh + (s0 + 16 + l16) * HD + 16 * g);
        v8f z = {};
        v8f s_0 = __builtin_amdgcn_wmma_f32_16x16x32_bf16(false, aq, false, bk0,
                                                          (short)0, z, false, false);
        v8f s_1 = __builtin_amdgcn_wmma_f32_16x16x32_bf16(false, aq, false, bk1,
                                                          (short)0, z, false, false);
#pragma unroll
        for (int r = 0; r < 8; ++r) {
            const int rowt = q0 + r + 8 * g;
            float x0 = s_0[r] * scale;
            float x1 = s_1[r] * scale;
            if (s0 + l16 > rowt)      x0 = -INFINITY;   // causal mask
            if (s0 + 16 + l16 > rowt) x1 = -INFINITY;
            float tmax = rowmax16(fmaxf(x0, x1));
            float mnew = fmaxf(mrow[r], tmax);
            float alpha = __expf(mrow[r] - mnew);
            float p0 = __expf(x0 - mnew);
            float p1 = __expf(x1 - mnew);
            lrow[r] = lrow[r] * alpha + rowsum16(p0 + p1);
            mrow[r] = mnew;
            O0[r] *= alpha;
            O1[r] *= alpha;
            Plds[(r + 8 * g) * 32 + l16]      = (bf16_t)p0;
            Plds[(r + 8 * g) * 32 + 16 + l16] = (bf16_t)p1;
        }
        __syncthreads();   // single-wave WG: cheap; orders LDS C->A relayout
        v16bf ap  = load_a_contig(&Plds[l16 * 32], g);
        // B = V tile: lane column d = (d0 + l16); 16 contiguous keys from V^T.
        v16bf bv0 = load_b_contig16(Vth + l16 * T_SEQ        + s0 + 16 * g);
        v16bf bv1 = load_b_contig16(Vth + (16 + l16) * T_SEQ + s0 + 16 * g);
        O0 = __builtin_amdgcn_wmma_f32_16x16x32_bf16(false, ap, false, bv0,
                                                     (short)0, O0, false, false);
        O1 = __builtin_amdgcn_wmma_f32_16x16x32_bf16(false, ap, false, bv1,
                                                     (short)0, O1, false, false);
        __syncthreads();
    }

    const int bn = bnh >> 3, h = bnh & 7;
#pragma unroll
    for (int r = 0; r < 8; ++r) {
        const int t = q0 + r + 8 * g;
        const float inv = 1.0f / lrow[r];
        bf16_t* dst = AO + (bn * T_SEQ + t) * E_DIM + h * HD;
        dst[l16]      = (bf16_t)(O0[r] * inv);
        dst[16 + l16] = (bf16_t)(O1[r] * inv);
    }
}

// ---- Stage 4: projection GEMM + bias + permute back to (B,T,N,E) fp32 ------
// Wt is w_proj transposed (256 x 256) -> contiguous B loads.
__global__ void proj_gemm_kernel(const bf16_t* __restrict__ A,
                                 const bf16_t* __restrict__ Wt,
                                 const float*  __restrict__ bias,
                                 float* __restrict__ out) {
    const int lane = threadIdx.x;
    const int l16  = lane & 15;
    const int g    = lane >> 4;
    const int tm   = blockIdx.x;   // 512
    const int tn   = blockIdx.y;   // 16
    const int col  = tn * 16 + l16;
    const bf16_t* arow = A  + (tm * 16 + l16) * E_DIM;
    const bf16_t* brow = Wt + col * E_DIM;
    v8f c = {};
#pragma unroll
    for (int kb = 0; kb < E_DIM; kb += 32) {
        v16bf a = load_a_contig(arow + kb, g);
        v16bf b = load_b_contig16(brow + kb + 16 * g);
        c = __builtin_amdgcn_wmma_f32_16x16x32_bf16(false, a, false, b,
                                                    (short)0, c, false, false);
    }
    const float bv = bias[col];
#pragma unroll
    for (int r = 0; r < 8; ++r) {
        const int rowM = tm * 16 + r + 8 * g;  // bn*T + t
        const int bn = rowM >> 10, t = rowM & 1023;
        const int b = bn >> 2,     n = bn & 3;
        out[((b * T_SEQ + t) * 4 + n) * E_DIM + col] = c[r] + bv;
    }
}

// ---------------------------------------------------------------------------
extern "C" void kernel_launch(void* const* d_in, const int* in_sizes, int n_in,
                              void* d_out, int out_size, void* d_ws, size_t ws_size,
                              hipStream_t stream) {
    const float* x_in   = (const float*)d_in[0];   // (2,1024,4,256) fp32
    const float* w_attn = (const float*)d_in[1];   // (256,768)
    const float* b_attn = (const float*)d_in[2];   // (768,)
    const float* w_proj = (const float*)d_in[3];   // (256,256)
    const float* b_proj = (const float*)d_in[4];   // (256,)
    float* out = (float*)d_out;

    // Workspace layout (~29 MB total; offsets MB-aligned):
    char* ws = (char*)d_ws;
    bf16_t* Xbf   = (bf16_t*)(ws);                              // 4 MB
    bf16_t* Wabf  = (bf16_t*)(ws + (size_t)(4  << 20));         // 384 KB (W_attn^T)
    bf16_t* Wpbf  = (bf16_t*)(ws + (size_t)(4  << 20) + (512 << 10)); // 128 KB (W_proj^T)
    bf16_t* QKVbf = (bf16_t*)(ws + (size_t)(5  << 20));         // 12 MB
    bf16_t* Qbf   = (bf16_t*)(ws + (size_t)(17 << 20));         // 4 MB
    bf16_t* Kbf   = (bf16_t*)(ws + (size_t)(21 << 20));         // 4 MB
    bf16_t* Vbf   = (bf16_t*)(ws + (size_t)(25 << 20));         // 4 MB (V^T per head)
    bf16_t* AObf  = (bf16_t*)(ws + (size_t)(5  << 20));         // reuses QKV region

    permute_input_kernel<<<8192, 256, 0, stream>>>(x_in, Xbf);
    transpose_to_bf16_kernel<<<768, 256, 0, stream>>>(w_attn, Wabf, E_DIM, QKV_N);
    transpose_to_bf16_kernel<<<256, 256, 0, stream>>>(w_proj, Wpbf, E_DIM, E_DIM);

    qkv_gemm_kernel<<<dim3(512, 48), 32, 0, stream>>>(Xbf, Wabf, b_attn, QKVbf);
    rope_split_kernel<<<4096, 256, 0, stream>>>(QKVbf, Qbf, Kbf, Vbf);
    attn_kernel<<<dim3(64, 64), 32, 0, stream>>>(Qbf, Kbf, Vbf, AObf);
    proj_gemm_kernel<<<dim3(512, 16), 32, 0, stream>>>(AObf, Wpbf, b_proj, out);
}
